// Combined_patch_attention_20761871909458
// MI455X (gfx1250) — matrix operand order repaired and checked by reading, not compiled, not run on hardware
//
#include <hip/hip_runtime.h>

// Problem constants (reference: B=4, H=16, P=1024, D=128)
#define B_ 4
#define H_ 16
#define P_ 1024
#define D_ 128
#define NELEM (B_ * H_ * P_ * D_)   // 8388608 per tensor

#define LOG2E 1.44269504088896340736f

typedef __attribute__((ext_vector_type(16))) _Float16 v16h;
typedef __attribute__((ext_vector_type(8)))  _Float16 v8h;
typedef __attribute__((ext_vector_type(8)))  float    v8f;

// Bare v_exp_f32 (2^x); softmax done in log2 domain so no log2e multiplies.
__device__ __forceinline__ float exp2_fast(float x) {
  return __builtin_amdgcn_exp2f(x);
}

// ---------------------------------------------------------------------------
// Pass 0a: y = f16( silu(x) * scale * log2e )  (q path; score scale and the
//          softmax log2e base-change folded into the A operand)
// Pass 0b: y = f16( silu(x) )                  (k path)
// ---------------------------------------------------------------------------
__global__ void silu_scale_to_f16(const float* __restrict__ x,
                                  _Float16* __restrict__ y,
                                  const float* __restrict__ scale_p, int n) {
  const float sc = scale_p[0] * LOG2E;
  int i = blockIdx.x * blockDim.x + threadIdx.x;
  int stride = gridDim.x * blockDim.x;
  for (; i < n; i += stride) {
    float v = x[i];
    y[i] = (_Float16)(sc * v / (1.0f + __expf(-v)));
  }
}

__global__ void silu_to_f16(const float* __restrict__ x,
                            _Float16* __restrict__ y, int n) {
  int i = blockIdx.x * blockDim.x + threadIdx.x;
  int stride = gridDim.x * blockDim.x;
  for (; i < n; i += stride) {
    float v = x[i];
    y[i] = (_Float16)(v / (1.0f + __expf(-v)));
  }
}

// ---------------------------------------------------------------------------
// Main kernel: one wave per (head, 32-row M-group) = two 16-row M-tiles.
//   Pass 1: per-lane online (max, sum 2^(s-max)) over 64 N-tiles; no
//           cross-lane ops in the hot loop. One butterfly merge at the end.
//   Pass 2: recompute scores, write 2^(s - max) * inv_sum.
// WMMA: v_wmma_f32_16x16x32_f16, K=128 via 4 chained steps, two independent
// accumulator chains (one per M-tile) sharing each B tile.
// ---------------------------------------------------------------------------
__global__ __launch_bounds__(128) void
attn_wmma_kernel(const _Float16* __restrict__ qh,
                 const _Float16* __restrict__ kh,
                 float* __restrict__ out) {
  const int lane  = threadIdx.x & 31;
  const int wid   = threadIdx.x >> 5;
  const int gw    = blockIdx.x * 4 + wid;   // 0..2047
  const int bh    = gw >> 5;                // head index 0..63
  const int m0    = (gw & 31) << 5;         // M-group base row (32 rows)
  const int nlane = lane & 15;
  const int half  = lane >> 4;              // 0: lanes 0-15, 1: lanes 16-31

  // --- A operands: 2 M-tiles x 4 K-chunks of silu(q)*scale*log2e -----------
  // A 16x32 as loaded here (contiguous k, same order as B): lane (l&15) holds row M; lanes<16 take K offsets
  // {0..15}, lanes>=16 take {16..31} per 32-K chunk.
  v16h amat[2][4];
#pragma unroll
  for (int i = 0; i < 2; ++i) {
    const _Float16* arow = qh + ((size_t)bh * P_ + m0 + i * 16 + nlane) * D_;
#pragma unroll
    for (int kk = 0; kk < 4; ++kk) {
      const int d0 = kk * 32;
      v8h lo = *(const v8h*)(arow + d0 + half * 16);
      v8h hi = *(const v8h*)(arow + d0 + 8 + half * 16);
      amat[i][kk] = __builtin_shufflevector(lo, hi, 0, 1, 2, 3, 4, 5, 6, 7,
                                            8, 9, 10, 11, 12, 13, 14, 15);
    }
  }

  // B 32x16 layout: lane (l&15) holds column N; each lane reads 16 contiguous
  // K values at offset half*16 within each 32-K chunk.
  const _Float16* kbase =
      kh + (size_t)bh * P_ * D_ + (size_t)nlane * D_ + half * 16;

  float mx[2][8], sm[2][8];
#pragma unroll
  for (int i = 0; i < 2; ++i)
#pragma unroll
    for (int r = 0; r < 8; ++r) { mx[i][r] = -3.0e38f; sm[i][r] = 0.0f; }

  // ---------------- Pass 1: per-lane online max / sum of 2^ ----------------
  for (int t = 0; t < 64; ++t) {
    const _Float16* brow = kbase + (size_t)(t * 16) * D_;
    v16h b[4];
#pragma unroll
    for (int kk = 0; kk < 4; ++kk) b[kk] = *(const v16h*)(brow + kk * 32);

    v8f c0 = {}, c1 = {};
#pragma unroll
    for (int kk = 0; kk < 4; ++kk)
      c0 = __builtin_amdgcn_wmma_f32_16x16x32_f16(false, amat[0][kk], false,
                                                  b[kk], (short)0, c0, false,
                                                  false);
#pragma unroll
    for (int kk = 0; kk < 4; ++kk)
      c1 = __builtin_amdgcn_wmma_f32_16x16x32_f16(false, amat[1][kk], false,
                                                  b[kk], (short)0, c1, false,
                                                  false);

#pragma unroll
    for (int r = 0; r < 8; ++r) {
      {
        float s  = c0[r];
        float nm = fmaxf(mx[0][r], s);
        sm[0][r] = sm[0][r] * exp2_fast(mx[0][r] - nm) + exp2_fast(s - nm);
        mx[0][r] = nm;
      }
      {
        float s  = c1[r];
        float nm = fmaxf(mx[1][r], s);
        sm[1][r] = sm[1][r] * exp2_fast(mx[1][r] - nm) + exp2_fast(s - nm);
        mx[1][r] = nm;
      }
    }
  }

  // ------------- One-time butterfly (max,sum) merge across 16 lanes --------
  float inv[2][8];
#pragma unroll
  for (int i = 0; i < 2; ++i)
#pragma unroll
    for (int r = 0; r < 8; ++r) {
#pragma unroll
      for (int st = 1; st < 16; st <<= 1) {
        float omx = __shfl_xor(mx[i][r], st, 32);
        float osm = __shfl_xor(sm[i][r], st, 32);
        float nm  = fmaxf(mx[i][r], omx);
        sm[i][r] = sm[i][r] * exp2_fast(mx[i][r] - nm) +
                   osm * exp2_fast(omx - nm);
        mx[i][r] = nm;
      }
      inv[i][r] = __builtin_amdgcn_rcpf(sm[i][r]);
    }

  // ---------------- Pass 2: recompute scores, write probabilities ----------
  // C/D layout: VGPR r, lanes 0-15 -> row +r, lanes 16-31 -> row +8+r.
  float* obase = out + ((size_t)bh * P_ + m0 + half * 8) * (size_t)P_ + nlane;
  for (int t = 0; t < 64; ++t) {
    const _Float16* brow = kbase + (size_t)(t * 16) * D_;
    v16h b[4];
#pragma unroll
    for (int kk = 0; kk < 4; ++kk) b[kk] = *(const v16h*)(brow + kk * 32);

    v8f c0 = {}, c1 = {};
#pragma unroll
    for (int kk = 0; kk < 4; ++kk)
      c0 = __builtin_amdgcn_wmma_f32_16x16x32_f16(false, amat[0][kk], false,
                                                  b[kk], (short)0, c0, false,
                                                  false);
#pragma unroll
    for (int kk = 0; kk < 4; ++kk)
      c1 = __builtin_amdgcn_wmma_f32_16x16x32_f16(false, amat[1][kk], false,
                                                  b[kk], (short)0, c1, false,
                                                  false);

    float* o = obase + t * 16;
#pragma unroll
    for (int r = 0; r < 8; ++r) {
      o[(size_t)(r)      * P_] = exp2_fast(c0[r] - mx[0][r]) * inv[0][r];
      o[(size_t)(16 + r) * P_] = exp2_fast(c1[r] - mx[1][r]) * inv[1][r];
    }
  }
}

// ---------------------------------------------------------------------------
// Host launch. d_ws layout: [0,16MB) silu(q)*scale*log2e f16, [16MB,32MB)
// silu(k) f16. Requires ws_size >= 32 MiB.
// ---------------------------------------------------------------------------
extern "C" void kernel_launch(void* const* d_in, const int* in_sizes, int n_in,
                              void* d_out, int out_size, void* d_ws,
                              size_t ws_size, hipStream_t stream) {
  (void)in_sizes; (void)n_in; (void)out_size; (void)ws_size;
  const float* q     = (const float*)d_in[0];
  const float* k     = (const float*)d_in[1];
  const float* scale = (const float*)d_in[2];
  float* out         = (float*)d_out;

  _Float16* qh = (_Float16*)d_ws;
  _Float16* kh = qh + (size_t)NELEM;

  // Pass 0: activations (scale * log2e folded into q) to f16 workspace.
  silu_scale_to_f16<<<4096, 256, 0, stream>>>(q, qh, scale, NELEM);
  silu_to_f16<<<4096, 256, 0, stream>>>(k, kh, NELEM);

  // Main attention kernel: 2048 waves = 512 blocks x 4 waves.
  attn_wmma_kernel<<<512, 128, 0, stream>>>(qh, kh, out);
}
